// AggregationMPNN_19868518711811
// MI455X (gfx1250) — matrix operand order repaired and checked by reading, not compiled, not run on hardware
//
#include <hip/hip_runtime.h>
#include <cstdint>
#include <cstddef>

// Problem constants (match reference)
#define B_    64
#define N_    64
#define E_    16
#define H_    128
#define M_    128
#define OUT_  128
#define PASSES 3
#define ROWS  (B_ * N_)          // 4096 node rows

typedef __attribute__((ext_vector_type(16))) _Float16 v16h;
typedef __attribute__((ext_vector_type(8)))  _Float16 v8h;
typedef __attribute__((ext_vector_type(8)))  float    v8f;

// ---------------------------------------------------------------------------
// WMMA helpers (CDNA5: v_wmma_f32_16x16x32_f16, wave32)
// ---------------------------------------------------------------------------
__device__ inline v8f wmma16(v16h a, v16h b, v8f c) {
    // (neg_a, A, neg_b, B, c_mod, C, reuse_a, reuse_b)
    return __builtin_amdgcn_wmma_f32_16x16x32_f16(
        false, a, false, b, (short)0, c, false, false);
}

// A fragment: 16x32 (MxK) from row-major f16 A (lda halves per row).
// Contiguous k order (B packed to match): lanes 0-15 -> M=lane, halves {0..15}->K k0+0..15
//             lanes 16-31 -> M=lane-16, halves {0..15}->K k0+16..31
// => per lane: two contiguous 16-byte loads.
__device__ inline v16h load_a_frag16(const _Float16* __restrict__ A, int lda,
                                     int row0, int k0, int lane) {
    int row = row0 + (lane & 15);
    const _Float16* p = A + (size_t)row * lda + k0 + ((lane & 16) ? 16 : 0);
    v8h lo = *(const v8h*)p;
    v8h hi = *(const v8h*)(p + 8);
    return __builtin_shufflevector(lo, hi, 0,1,2,3,4,5,6,7,8,9,10,11,12,13,14,15);
}

// B fragment from pre-packed fragment-major buffer: [tile][lane][16 halves].
// One aligned 32-byte vector load per lane.
__device__ inline v16h load_b_packed(const _Float16* __restrict__ frags,
                                     int tileIdx, int lane) {
    return *(const v16h*)(frags + ((size_t)tileIdx * 32 + lane) * 16);
}

// C/D 16x16 f32: VGPR r -> M = r (lanes 0-15) or r+8 (lanes 16-31), N = lane&15.
__device__ inline void store_c(float* __restrict__ C, int ldc, int row0, int c0,
                               int lane, v8f acc, float bias) {
    int col   = c0 + (lane & 15);
    int rbase = row0 + ((lane & 16) ? 8 : 0);
#pragma unroll
    for (int r = 0; r < 8; ++r)
        C[(size_t)(rbase + r) * ldc + col] = acc[r] + bias;
}

// ---------------------------------------------------------------------------
// One-time weight pre-pack: f32 row-major (K x n) -> f16 WMMA fragment layout.
// grid = (K/32, n/16), block = 32.  Layout matches load_b_packed:
// element B[k][n]: lane = (n&15) + 16*((k&31) >= 16), half = k&15.
// ---------------------------------------------------------------------------
__global__ void mpnn_pack_b(const float* __restrict__ W, _Float16* __restrict__ out, int n) {
    int kt = blockIdx.x, ct = blockIdx.y, lane = threadIdx.x;
    int nct = gridDim.y;
    int col = ct * 16 + (lane & 15);
    int kb  = kt * 32 + ((lane & 16) ? 16 : 0);
    _Float16* dst = out + ((size_t)(kt * nct + ct) * 32 + lane) * 16;
#pragma unroll
    for (int h = 0; h < 16; ++h)
        dst[h] = (_Float16)W[(size_t)(kb + h) * n + col];
}

// ---------------------------------------------------------------------------
// Preprocess: dense one-hot edges (B,N,N,E) -> etype (u8) + 64-bit row masks
// ---------------------------------------------------------------------------
__global__ void mpnn_preprocess(const float* __restrict__ edges,
                                unsigned char* __restrict__ etype,
                                unsigned long long* __restrict__ rowmask) {
    __shared__ unsigned int smask[2];
    int row = blockIdx.x;            // b*N + i
    int j   = threadIdx.x;           // 0..63
    if (j < 2) smask[j] = 0u;
    __syncthreads();
    const float* ep = edges + ((size_t)row * N_ + j) * E_;
    int et = 0; bool exists = false;
#pragma unroll
    for (int e = 0; e < E_; ++e)
        if (ep[e] > 0.5f) { et = e; exists = true; }
    etype[(size_t)row * N_ + j] = (unsigned char)et;
    if (exists) atomicOr(&smask[j >> 5], 1u << (j & 31));
    __syncthreads();
    if (j == 0)
        rowmask[row] = ((unsigned long long)smask[1] << 32) | (unsigned long long)smask[0];
}

// Edge-type contribution tables: tab[e][m] = W[H+e][m] + b[m]  (bias folded in)
__global__ void mpnn_tabs(const float* __restrict__ Watt, const float* __restrict__ batt,
                          const float* __restrict__ Wmsg, const float* __restrict__ bmsg,
                          float* __restrict__ att_tab, float* __restrict__ msg_tab) {
    int idx = blockIdx.x * blockDim.x + threadIdx.x;
    if (idx >= E_ * M_) return;
    int e = idx >> 7, m = idx & (M_ - 1);
    att_tab[idx] = Watt[(size_t)(H_ + e) * M_ + m] + batt[m];
    msg_tab[idx] = Wmsg[(size_t)(H_ + e) * M_ + m] + bmsg[m];
}

// Init: hidden <- nodes (f32) + f16 shadows of both
__global__ void mpnn_init(const float* __restrict__ nodes, float* __restrict__ hidden,
                          _Float16* __restrict__ hidden_h, _Float16* __restrict__ nodes_h) {
    int idx = blockIdx.x * blockDim.x + threadIdx.x;
    float v = nodes[idx];
    hidden[idx]   = v;
    _Float16 h = (_Float16)v;
    hidden_h[idx] = h;
    nodes_h[idx]  = h;
}

// ---------------------------------------------------------------------------
// Per-pass projections: Hatt = hidden @ W_att[:H], Hmsg = hidden @ W_msg[:H]
// One wave -> one 16x16 tile of both outputs (A fragment reused).
// ---------------------------------------------------------------------------
__global__ void mpnn_hh_gemm(const _Float16* __restrict__ hidden_h,
                             const _Float16* __restrict__ WattP,
                             const _Float16* __restrict__ WmsgP,
                             float* __restrict__ Hatt, float* __restrict__ Hmsg) {
    int lane = threadIdx.x & 31, wave = threadIdx.x >> 5;
    int row0 = blockIdx.x * 16;
    int ct   = blockIdx.y * 4 + wave;           // 0..7 (nct = 8)
    v8f ca = {}, cm = {};
#pragma unroll
    for (int kt = 0; kt < H_ / 32; ++kt) {
        v16h a = load_a_frag16(hidden_h, H_, row0, kt * 32, lane);
        ca = wmma16(a, load_b_packed(WattP, kt * 8 + ct, lane), ca);
        cm = wmma16(a, load_b_packed(WmsgP, kt * 8 + ct, lane), cm);
    }
    store_c(Hatt, M_, row0, ct * 16, lane, ca, 0.f);
    store_c(Hmsg, M_, row0, ct * 16, lane, cm, 0.f);
}

// GRU gate GEMM: C(4096 x 384) = A(4096 x 128) @ B(128 x 384) + bias (nct = 24)
__global__ void mpnn_gemm_bias(const _Float16* __restrict__ A,
                               const _Float16* __restrict__ Bfr,
                               const float* __restrict__ bias, float* __restrict__ C) {
    int lane = threadIdx.x & 31, wave = threadIdx.x >> 5;
    int row0 = blockIdx.x * 16;
    int ct   = blockIdx.y * 4 + wave;           // 0..23
    v8f acc = {};
#pragma unroll
    for (int kt = 0; kt < H_ / 32; ++kt) {
        v16h a = load_a_frag16(A, H_, row0, kt * 32, lane);
        acc = wmma16(a, load_b_packed(Bfr, kt * 24 + ct, lane), acc);
    }
    store_c(C, 3 * H_, row0, ct * 16, lane, acc, bias[ct * 16 + (lane & 15)]);
}

// ---------------------------------------------------------------------------
// Attention + message aggregation. Block = one (b,i) row, thread = channel m.
// Online softmax over neighbor set bits; per-channel softmax as in reference.
// Writes messages directly as the f16 A-operand for the following GEMM.
// ---------------------------------------------------------------------------
__global__ void mpnn_attmsg(const float* __restrict__ Hatt, const float* __restrict__ Hmsg,
                            const float* __restrict__ att_tab, const float* __restrict__ msg_tab,
                            const unsigned char* __restrict__ etype,
                            const unsigned long long* __restrict__ rowmask,
                            _Float16* __restrict__ messages_h) {
    int row = blockIdx.x;            // b*N + i
    int m   = threadIdx.x;           // 0..127
    int b   = row >> 6;
    unsigned long long mask = rowmask[row];
    float mx = -3.0e38f, sum = 0.f, acc = 0.f;
    unsigned long long mm = mask;
    while (mm) {
        int j = __builtin_ctzll(mm);
        mm &= mm - 1;
        int et = etype[(size_t)row * N_ + j];
        size_t nb = (size_t)(b * N_ + j) * M_;
        float e = Hatt[nb + m] + att_tab[et * M_ + m];
        float v = Hmsg[nb + m] + msg_tab[et * M_ + m];
        float nm   = fmaxf(mx, e);
        float corr = __expf(mx - nm);
        float w    = __expf(e - nm);
        sum = sum * corr + w;
        acc = acc * corr + w * v;
        mx  = nm;
    }
    messages_h[(size_t)row * M_ + m] = (_Float16)(mask ? (acc / sum) : 0.f);
}

// GRU cell update, masked by node_mask (rowmask != 0); refreshes f16 shadow
__global__ void mpnn_gru_update(const float* __restrict__ gi, const float* __restrict__ gh,
                                const unsigned long long* __restrict__ rowmask,
                                float* __restrict__ hidden, _Float16* __restrict__ hidden_h) {
    int idx = blockIdx.x * blockDim.x + threadIdx.x;   // < ROWS*H_
    int row = idx >> 7;
    size_t g = (size_t)row * (3 * H_) + (idx & (H_ - 1));
    float h  = hidden[idx];
    float r  = 1.f / (1.f + __expf(-(gi[g] + gh[g])));
    float z  = 1.f / (1.f + __expf(-(gi[g + H_] + gh[g + H_])));
    float n  = tanhf(gi[g + 2 * H_] + r * gh[g + 2 * H_]);
    float hn = (1.f - z) * n + z * h;
    float hv = rowmask[row] ? hn : h;
    hidden[idx]   = hv;
    hidden_h[idx] = (_Float16)hv;
}

// ---------------------------------------------------------------------------
// Readout GEMMs: gate = sigmoid([hidden|nodes] @ W_gate + b_gate),
//                emb  = hidden @ W_emb + b_emb.  A fragment reuse on hidden.
// WgateP packed with 8 k-tiles (K=256): kt 0..3 -> hidden, kt 4..7 -> nodes.
// ---------------------------------------------------------------------------
__global__ void mpnn_readout_gemm(const _Float16* __restrict__ hidden_h,
                                  const _Float16* __restrict__ nodes_h,
                                  const _Float16* __restrict__ WgateP, const float* __restrict__ bgate,
                                  const _Float16* __restrict__ WembP,  const float* __restrict__ bemb,
                                  float* __restrict__ gate, float* __restrict__ emb) {
    int lane = threadIdx.x & 31, wave = threadIdx.x >> 5;
    int row0 = blockIdx.x * 16;
    int ct   = blockIdx.y * 4 + wave;           // 0..7
    v8f cg = {}, ce = {};
#pragma unroll
    for (int kt = 0; kt < 4; ++kt) {
        v16h a = load_a_frag16(hidden_h, H_, row0, kt * 32, lane);
        cg = wmma16(a, load_b_packed(WgateP, kt * 8 + ct, lane), cg);
        ce = wmma16(a, load_b_packed(WembP,  kt * 8 + ct, lane), ce);
    }
#pragma unroll
    for (int kt = 0; kt < 4; ++kt) {            // concat tail: nodes x W_gate[H:]
        v16h a = load_a_frag16(nodes_h, H_, row0, kt * 32, lane);
        cg = wmma16(a, load_b_packed(WgateP, (kt + 4) * 8 + ct, lane), cg);
    }
    int col = ct * 16 + (lane & 15);
    int rb  = row0 + ((lane & 16) ? 8 : 0);
    float bg = bgate[col], be = bemb[col];
#pragma unroll
    for (int r = 0; r < 8; ++r) {
        size_t o = (size_t)(rb + r) * OUT_ + col;
        gate[o] = 1.f / (1.f + __expf(-(cg[r] + bg)));
        emb[o]  = ce[r] + be;
    }
}

// Graph-gather: out[b,o] = sum_i node_mask * gate * emb
__global__ void mpnn_reduce(const float* __restrict__ gate, const float* __restrict__ emb,
                            const unsigned long long* __restrict__ rowmask,
                            float* __restrict__ out) {
    int b = blockIdx.x, o = threadIdx.x;
    float acc = 0.f;
#pragma unroll 4
    for (int i = 0; i < N_; ++i) {
        int row = b * N_ + i;
        if (rowmask[row]) {
            size_t p = (size_t)row * OUT_ + o;
            acc += gate[p] * emb[p];
        }
    }
    out[(size_t)b * OUT_ + o] = acc;
}

// ---------------------------------------------------------------------------
extern "C" void kernel_launch(void* const* d_in, const int* in_sizes, int n_in,
                              void* d_out, int out_size, void* d_ws, size_t ws_size,
                              hipStream_t stream) {
    (void)in_sizes; (void)n_in; (void)out_size; (void)ws_size;
    const float* nodes = (const float*)d_in[0];
    const float* edges = (const float*)d_in[1];
    const float* Watt  = (const float*)d_in[2];
    const float* batt  = (const float*)d_in[3];
    const float* Wmsg  = (const float*)d_in[4];
    const float* bmsg  = (const float*)d_in[5];
    const float* W_i   = (const float*)d_in[6];
    const float* b_i   = (const float*)d_in[7];
    const float* W_h   = (const float*)d_in[8];
    const float* b_h   = (const float*)d_in[9];
    const float* Wgate = (const float*)d_in[10];
    const float* bgate = (const float*)d_in[11];
    const float* Wemb  = (const float*)d_in[12];
    const float* bemb  = (const float*)d_in[13];
    float* out = (float*)d_out;

    // Workspace carve-up (256B aligned chunks)
    char* ws = (char*)d_ws;
    size_t off = 0;
    auto alloc = [&](size_t bytes) { void* p = ws + off; off = (off + bytes + 255) & ~(size_t)255; return p; };
    float*     hidden     = (float*)alloc((size_t)ROWS * H_ * 4);
    _Float16*  hidden_h   = (_Float16*)alloc((size_t)ROWS * H_ * 2);
    _Float16*  nodes_h    = (_Float16*)alloc((size_t)ROWS * H_ * 2);
    _Float16*  messages_h = (_Float16*)alloc((size_t)ROWS * M_ * 2);
    float*     Hatt       = (float*)alloc((size_t)ROWS * M_ * 4);
    float*     Hmsg       = (float*)alloc((size_t)ROWS * M_ * 4);
    float*     gi         = (float*)alloc((size_t)ROWS * 3 * H_ * 4);
    float*     gh         = (float*)alloc((size_t)ROWS * 3 * H_ * 4);
    float*     gateB      = (float*)alloc((size_t)ROWS * OUT_ * 4);
    float*     embB       = (float*)alloc((size_t)ROWS * OUT_ * 4);
    float*     att_tab    = (float*)alloc((size_t)E_ * M_ * 4);
    float*     msg_tab    = (float*)alloc((size_t)E_ * M_ * 4);
    _Float16*  WattP      = (_Float16*)alloc((size_t)H_ * M_ * 2);
    _Float16*  WmsgP      = (_Float16*)alloc((size_t)H_ * M_ * 2);
    _Float16*  WiP        = (_Float16*)alloc((size_t)H_ * 3 * H_ * 2);
    _Float16*  WhP        = (_Float16*)alloc((size_t)H_ * 3 * H_ * 2);
    _Float16*  WgateP     = (_Float16*)alloc((size_t)2 * H_ * OUT_ * 2);
    _Float16*  WembP      = (_Float16*)alloc((size_t)H_ * OUT_ * 2);
    unsigned long long* rowmask = (unsigned long long*)alloc((size_t)ROWS * 8);
    unsigned char*      etype   = (unsigned char*)alloc((size_t)ROWS * N_);

    // One-time: structure extraction, bias-folded edge tables, weight pre-pack
    mpnn_preprocess<<<ROWS, 64, 0, stream>>>(edges, etype, rowmask);
    mpnn_tabs<<<(E_ * M_ + 127) / 128, 128, 0, stream>>>(Watt, batt, Wmsg, bmsg, att_tab, msg_tab);
    mpnn_pack_b<<<dim3(H_ / 32, M_ / 16),        32, 0, stream>>>(Watt,  WattP,  M_);      // hidden rows only
    mpnn_pack_b<<<dim3(H_ / 32, M_ / 16),        32, 0, stream>>>(Wmsg,  WmsgP,  M_);
    mpnn_pack_b<<<dim3(H_ / 32, (3 * H_) / 16),  32, 0, stream>>>(W_i,   WiP,    3 * H_);
    mpnn_pack_b<<<dim3(H_ / 32, (3 * H_) / 16),  32, 0, stream>>>(W_h,   WhP,    3 * H_);
    mpnn_pack_b<<<dim3((2 * H_) / 32, OUT_ / 16),32, 0, stream>>>(Wgate, WgateP, OUT_);
    mpnn_pack_b<<<dim3(H_ / 32, OUT_ / 16),      32, 0, stream>>>(Wemb,  WembP,  OUT_);
    mpnn_init<<<(ROWS * H_) / 256, 256, 0, stream>>>(nodes, hidden, hidden_h, nodes_h);

    for (int p = 0; p < PASSES; ++p) {
        mpnn_hh_gemm<<<dim3(ROWS / 16, M_ / 64), 128, 0, stream>>>(hidden_h, WattP, WmsgP, Hatt, Hmsg);
        mpnn_attmsg<<<ROWS, M_, 0, stream>>>(Hatt, Hmsg, att_tab, msg_tab, etype, rowmask, messages_h);
        mpnn_gemm_bias<<<dim3(ROWS / 16, (3 * H_) / 64), 128, 0, stream>>>(messages_h, WiP, b_i, gi);
        mpnn_gemm_bias<<<dim3(ROWS / 16, (3 * H_) / 64), 128, 0, stream>>>(hidden_h,   WhP, b_h, gh);
        mpnn_gru_update<<<(ROWS * H_) / 256, 256, 0, stream>>>(gi, gh, rowmask, hidden, hidden_h);
    }

    mpnn_readout_gemm<<<dim3(ROWS / 16, OUT_ / 64), 128, 0, stream>>>(
        hidden_h, nodes_h, WgateP, bgate, WembP, bemb, gateB, embB);
    mpnn_reduce<<<B_, OUT_, 0, stream>>>(gateB, embB, rowmask, out);
}
